// ContextClassifier_57380763074836
// MI455X (gfx1250) — matrix operand order repaired and checked by reading, not compiled, not run on hardware
//
#include <hip/hip_runtime.h>
#include <hip/hip_bf16.h>
#include <math.h>

// ---------- types ----------
typedef __bf16 bf16;
typedef __attribute__((ext_vector_type(16))) __bf16 v16bf;
typedef __attribute__((ext_vector_type(8)))  __bf16 v8bf;
typedef __attribute__((ext_vector_type(8)))  float  v8f;
typedef __attribute__((ext_vector_type(4)))  unsigned int u32x4;
typedef __attribute__((ext_vector_type(8)))  int i32x8;

// ---------- problem constants ----------
#define T_SEQ   512
#define BATCH   32
#define H2      1024          // 2*H
#define DDIM    512
#define VOCAB   32000
#define LROWS   2048
#define NROWS   4096          // 2L
#define NSLICE  25
#define SLICE_COLS  (VOCAB / NSLICE)    // 1280
#define SLICE_TILES (SLICE_COLS / 16)   // 80
#define SLICE_PAIRS (SLICE_TILES / 2)   // 40
#define BP_M    64            // rows per block in big GEMM
#define BP_LD   520           // LDS row stride (bf16): 1024B row + 16B skew

// Use TDM (tensor_load_to_lds) for the feats->LDS staging in k_bigpart.
#define USE_TDM 1

// ---------- WMMA helpers ----------
__device__ __forceinline__ v8f wmma_bf16(v16bf a, v16bf b, v8f c) {
    return __builtin_amdgcn_wmma_f32_16x16x32_bf16(
        false, a, false, b, (short)0, c, false, false);
}

// A-matrix 16x32 bf16 fragment, row-major source with stride ld (bf16 elems).
// Lane l<16: row=l, K in {0..7} then {16..23}; lane>=16: row=l-16, K {8..15},{24..31}.
__device__ __forceinline__ v16bf load_a_frag(const bf16* tile, int ld) {
    const int lane = threadIdx.x & 31;
    const int r  = lane & 15;
    const int ko = (lane >> 4) << 3;           // 0 or 8
    const bf16* p = tile + (size_t)r * ld + ko;
    v8bf a = *(const v8bf*)(p);
    v8bf b = *(const v8bf*)(p + 16);
    v16bf f;
#pragma unroll
    for (int i = 0; i < 8; ++i) { f[i] = a[i]; f[i + 8] = b[i]; }
    return f;
}

// B-matrix 32x16 bf16 fragment. We compute C = A * W^T, so W's row n supplies
// B's column n. Lane l<16: col=l, K {0..7},{16..23}; lane>=16: col=l-16, K {8..15},{24..31}.
__device__ __forceinline__ v16bf load_b_frag(const bf16* wrows, int ld) {
    const int lane = threadIdx.x & 31;
    const int c  = lane & 15;
    const int kb = (lane >> 4) << 3;           // 0 or 8
    const bf16* p = wrows + (size_t)c * ld + kb;
    v8bf a = *(const v8bf*)(p);
    v8bf b = *(const v8bf*)(p + 16);
    v16bf f;
#pragma unroll
    for (int i = 0; i < 8; ++i) { f[i] = a[i]; f[i + 8] = b[i]; }
    return f;
}

__device__ __forceinline__ void split_store(float v, bf16* hi, bf16* lo, size_t i) {
    bf16 h = (bf16)v;
    hi[i] = h;
    lo[i] = (bf16)(v - (float)h);
}

// ---------- kernel 1: gather span features -> bf16 hi/lo [4096 x 1024] ----------
__global__ __launch_bounds__(256) void k_gather(
    const float* __restrict__ hidden, const int* __restrict__ begins,
    const int* __restrict__ ends, const int* __restrict__ bids,
    bf16* __restrict__ gh, bf16* __restrict__ gl)
{
    const int row = blockIdx.x;            // 0..4095
    const int tid = threadIdx.x;
    if (row < LROWS) {                     // context rows
        const int l = row;
        const int b = bids[l];
        const float* p0 = hidden + ((size_t)(begins[l] - 1) * BATCH + b) * H2;
        const float* p1 = hidden + ((size_t)(ends[l])       * BATCH + b) * H2;
        for (int c = tid; c < H2; c += 256) {
            float v = (c < (H2 / 2)) ? p0[c] : p1[c];
            split_store(v, gh, gl, (size_t)row * H2 + c);
        }
    } else {                               // phrase rows
        const int l = row - LROWS;
        const int b = bids[l];
        const float* p0 = hidden + ((size_t)(begins[l])     * BATCH + b) * H2;
        const float* p1 = hidden + ((size_t)(ends[l] - 1)   * BATCH + b) * H2;
        for (int c = tid; c < H2; c += 256) {
            float v = 0.5f * (p0[c] + p1[c]);
            split_store(v, gh, gl, (size_t)row * H2 + c);
        }
    }
}

// ---------- kernel 2: fp32 -> bf16 hi/lo split ----------
__global__ __launch_bounds__(256) void k_cvt(
    const float* __restrict__ src, bf16* __restrict__ hi, bf16* __restrict__ lo, int n)
{
    int i = blockIdx.x * 256 + threadIdx.x;
    if (i < n) split_store(src[i], hi, lo, (size_t)i);
}

// ---------- kernel 3: FFN GEMM + tanh (WMMA, hi/lo split) ----------
__global__ __launch_bounds__(256) void k_ffn(
    const bf16* __restrict__ gh, const bf16* __restrict__ gl,
    const bf16* __restrict__ wch, const bf16* __restrict__ wcl,
    const bf16* __restrict__ wph, const bf16* __restrict__ wpl,
    const float* __restrict__ bc, const float* __restrict__ bp,
    float* __restrict__ feats, bf16* __restrict__ fh, bf16* __restrict__ fl)
{
    const int rowBase = blockIdx.x * 32;   // 128 blocks
    const bool ctx = rowBase < LROWS;
    const bf16* Wh = ctx ? wch : wph;
    const bf16* Wl = ctx ? wcl : wpl;
    const float* bias = ctx ? bc : bp;
    const int wave = threadIdx.x >> 5;
    const int lane = threadIdx.x & 31;
    const int half = lane >> 4;

    for (int t = 0; t < 4; ++t) {          // 8 waves x 4 tiles = 512 cols
        const int col0 = (wave * 4 + t) * 16;
        v8f acc[2];
#pragma unroll
        for (int s = 0; s < 2; ++s)
#pragma unroll
            for (int j = 0; j < 8; ++j) acc[s][j] = 0.f;

        for (int k = 0; k < H2; k += 32) {
            v16bf bh = load_b_frag(Wh + (size_t)col0 * H2 + k, H2);
            v16bf bl = load_b_frag(Wl + (size_t)col0 * H2 + k, H2);
#pragma unroll
            for (int s = 0; s < 2; ++s) {
                const size_t abase = (size_t)(rowBase + s * 16) * H2 + k;
                v16bf ah = load_a_frag(gh + abase, H2);
                v16bf al = load_a_frag(gl + abase, H2);
                acc[s] = wmma_bf16(ah, bh, acc[s]);
                acc[s] = wmma_bf16(ah, bl, acc[s]);
                acc[s] = wmma_bf16(al, bh, acc[s]);
            }
        }
        const int c = col0 + (lane & 15);
        const float bv = bias[c];
#pragma unroll
        for (int s = 0; s < 2; ++s)
#pragma unroll
            for (int j = 0; j < 8; ++j) {
                const int r = rowBase + s * 16 + half * 8 + j;
                float v = tanhf(acc[s][j] + bv);
                feats[(size_t)r * DDIM + c] = v;
                split_store(v, fh, fl, (size_t)r * DDIM + c);
            }
    }
}

// ---------- kernel 4: exact fp32 tag logits ----------
__global__ __launch_bounds__(64) void k_taglogits(
    const float* __restrict__ feats, const float* __restrict__ Wo,
    const float* __restrict__ bo, const int* __restrict__ tags,
    float* __restrict__ tagLogit)
{
    const int row = blockIdx.x;            // 0..4095
    const int wv  = threadIdx.x >> 5;      // tag index 0/1
    const int lane = threadIdx.x & 31;
    const int tg = tags[(row & (LROWS - 1)) * 2 + wv];
    const float* a = feats + (size_t)row * DDIM;
    const float* w = Wo + (size_t)tg * DDIM;
    float s = 0.f;
    for (int c = lane; c < DDIM; c += 32) s = fmaf(a[c], w[c], s);
#pragma unroll
    for (int off = 16; off; off >>= 1) s += __shfl_xor(s, off, 32);
    if (lane == 0) tagLogit[row * 2 + wv] = s + bo[tg];
}

// ---------- TDM: 2D tile (DDIM x BP_M, 2B elems) global -> LDS with row pad ----
#if USE_TDM
__device__ __forceinline__ void tdm_load_tile(const bf16* gsrc, bf16* lds_dst) {
    unsigned long long ga = (unsigned long long)(size_t)gsrc;
    unsigned lds = (unsigned)(size_t)(__attribute__((address_space(3))) bf16*)lds_dst;
    u32x4 g0;
    g0[0] = 1u;                                   // count=1, user mode, no gather
    g0[1] = lds;                                  // lds_addr (bytes)
    g0[2] = (unsigned)(ga & 0xffffffffu);         // global_addr[31:0]
    g0[3] = (unsigned)((ga >> 32) & 0x01ffffffu)  // global_addr[56:32]
          | (2u << 30);                           // type = 2 (image)
    i32x8 g1;
    g1[0] = (1 << 16)      // data_size = 1 -> 2 bytes
          | (1 << 20)      // pad_enable
          | (7 << 22)      // pad_interval code 7 -> every 256 DWORDs (1024 B)
          | (3 << 25);     // pad_amount  code 3 -> 4 DWORDs (16 B)
    g1[1] = (DDIM & 0xffff) << 16;                // tensor_dim0[15:0] = 512
    g1[2] = (DDIM >> 16) | ((BP_M & 0xffff) << 16); // dim0 hi | tensor_dim1 lo = 64
    g1[3] = (BP_M >> 16) | (DDIM << 16);          // dim1 hi | tile_dim0 = 512
    g1[4] = BP_M;                                 // tile_dim1 = 64, tile_dim2 = 0
    g1[5] = DDIM;                                 // tensor_dim0_stride[31:0] = 512
    g1[6] = 0;                                    // stride0 hi | stride1 lo
    g1[7] = 0;                                    // stride1 hi
    asm volatile("tensor_load_to_lds %0, %1" :: "s"(g0), "s"(g1) : "memory");
}
#endif

// ---------- kernel 5: big GEMM + fused online softmax partials ----------
// grid: (64 row tiles of 64, 25 vocab slices of 1280). partials[slice][row]={m,s}
__global__ __launch_bounds__(256) void k_bigpart(
    const bf16* __restrict__ fh, const bf16* __restrict__ fl,
    const bf16* __restrict__ wh, const bf16* __restrict__ wl,
    const float* __restrict__ bo, float* __restrict__ partials)
{
    __shared__ bf16 Ah[BP_M * BP_LD];
    __shared__ bf16 Al[BP_M * BP_LD];
    __shared__ float wpart[8][BP_M][2];

    const int rowBase = blockIdx.x * BP_M;
    const int slice   = blockIdx.y;
    const int colBase = slice * SLICE_COLS;
    const int tid  = threadIdx.x;
    const int wave = tid >> 5;
    const int lane = tid & 31;

#if USE_TDM
    if (wave == 0) {
        tdm_load_tile(fh + (size_t)rowBase * DDIM, Ah);
        tdm_load_tile(fl + (size_t)rowBase * DDIM, Al);
        __builtin_amdgcn_s_wait_tensorcnt(0);
    }
#else
    for (int idx = tid; idx < BP_M * (DDIM / 8); idx += 256) {
        const int r  = idx >> 6;
        const int c8 = idx & 63;
        *(v8bf*)(&Ah[r * BP_LD + c8 * 8]) = *(const v8bf*)(&fh[((size_t)(rowBase + r)) * DDIM + c8 * 8]);
        *(v8bf*)(&Al[r * BP_LD + c8 * 8]) = *(const v8bf*)(&fl[((size_t)(rowBase + r)) * DDIM + c8 * 8]);
    }
#endif
    __syncthreads();

    float m_run[8][8], s_run[8][8];        // [tile pair slot: 2 tiles x 4 subtiles]
#pragma unroll
    for (int s = 0; s < 8; ++s)
#pragma unroll
        for (int j = 0; j < 8; ++j) { m_run[s][j] = -3.0e38f; s_run[s][j] = 0.f; }

    for (int p = wave; p < SLICE_PAIRS; p += 8) {   // 5 pairs per wave
        const int col0 = colBase + p * 32;
        const bf16* Bh0 = wh + (size_t)col0 * DDIM;
        const bf16* Bl0 = wl + (size_t)col0 * DDIM;
        const bf16* Bh1 = Bh0 + (size_t)16 * DDIM;
        const bf16* Bl1 = Bl0 + (size_t)16 * DDIM;
        __builtin_prefetch(Bh0 + (size_t)8 * 32 * DDIM, 0, 0);  // next pair
        __builtin_prefetch(Bl0 + (size_t)8 * 32 * DDIM, 0, 0);

        v8f acc[8];
#pragma unroll
        for (int s = 0; s < 8; ++s)
#pragma unroll
            for (int j = 0; j < 8; ++j) acc[s][j] = 0.f;

        for (int k = 0; k < DDIM; k += 32) {
            v16bf bh0 = load_b_frag(Bh0 + k, DDIM);
            v16bf bl0 = load_b_frag(Bl0 + k, DDIM);
            v16bf bh1 = load_b_frag(Bh1 + k, DDIM);
            v16bf bl1 = load_b_frag(Bl1 + k, DDIM);
#pragma unroll
            for (int s = 0; s < 4; ++s) {
                v16bf ah = load_a_frag(&Ah[(s * 16) * BP_LD + k], BP_LD);
                v16bf al = load_a_frag(&Al[(s * 16) * BP_LD + k], BP_LD);
                acc[s]     = wmma_bf16(ah, bh0, acc[s]);
                acc[s]     = wmma_bf16(ah, bl0, acc[s]);
                acc[s]     = wmma_bf16(al, bh0, acc[s]);
                acc[s + 4] = wmma_bf16(ah, bh1, acc[s + 4]);
                acc[s + 4] = wmma_bf16(ah, bl1, acc[s + 4]);
                acc[s + 4] = wmma_bf16(al, bh1, acc[s + 4]);
            }
        }
        const float bias0 = bo[col0 + (lane & 15)];
        const float bias1 = bo[col0 + 16 + (lane & 15)];
#pragma unroll
        for (int s = 0; s < 8; ++s) {
            const float bias = (s < 4) ? bias0 : bias1;
#pragma unroll
            for (int j = 0; j < 8; ++j) {
                float v  = acc[s][j] + bias;
                float mo = m_run[s][j];
                float mn = fmaxf(mo, v);
                s_run[s][j] = s_run[s][j] * __expf(mo - mn) + __expf(v - mn);
                m_run[s][j] = mn;
            }
        }
    }

    // fold the two tile slots (same rows, different columns)
#pragma unroll
    for (int s = 0; s < 4; ++s)
#pragma unroll
        for (int j = 0; j < 8; ++j) {
            float m1 = m_run[s][j],     s1 = s_run[s][j];
            float m2 = m_run[s + 4][j], s2 = s_run[s + 4][j];
            float mn = fmaxf(m1, m2);
            s_run[s][j] = s1 * __expf(m1 - mn) + s2 * __expf(m2 - mn);
            m_run[s][j] = mn;
        }

    // reduce (m,s) across the 16 column-lanes of each half-wave
#pragma unroll
    for (int s = 0; s < 4; ++s)
#pragma unroll
        for (int j = 0; j < 8; ++j) {
            float m = m_run[s][j], sv = s_run[s][j];
#pragma unroll
            for (int off = 1; off < 16; off <<= 1) {
                float m2 = __shfl_xor(m, off, 32);
                float s2 = __shfl_xor(sv, off, 32);
                float mn = fmaxf(m, m2);
                sv = sv * __expf(m - mn) + s2 * __expf(m2 - mn);
                m = mn;
            }
            m_run[s][j] = m; s_run[s][j] = sv;
        }
    if ((lane & 15) == 0) {
        const int half = lane >> 4;
#pragma unroll
        for (int s = 0; s < 4; ++s)
#pragma unroll
            for (int j = 0; j < 8; ++j) {
                const int r = s * 16 + half * 8 + j;
                wpart[wave][r][0] = m_run[s][j];
                wpart[wave][r][1] = s_run[s][j];
            }
    }
    __syncthreads();
    if (tid < BP_M) {
        float m = -3.0e38f, sv = 0.f;
#pragma unroll
        for (int w = 0; w < 8; ++w) {
            float m2 = wpart[w][tid][0], s2 = wpart[w][tid][1];
            float mn = fmaxf(m, m2);
            sv = sv * __expf(m - mn) + s2 * __expf(m2 - mn);
            m = mn;
        }
        const size_t o = ((size_t)slice * NROWS + rowBase + tid) * 2;
        partials[o]     = m;
        partials[o + 1] = sv;
    }
}

// ---------- kernel 6: combine slices, loss, mean ----------
__global__ __launch_bounds__(256) void k_final(
    const float* __restrict__ partials, const float* __restrict__ tagLogit,
    const int* __restrict__ tags, const float* __restrict__ discard,
    float* __restrict__ out)
{
    __shared__ float red[256];
    float acc = 0.f;
    for (int r = threadIdx.x; r < NROWS; r += 256) {
        float m = -3.0e38f, sv = 0.f;
        for (int sl = 0; sl < NSLICE; ++sl) {
            const size_t o = ((size_t)sl * NROWS + r) * 2;
            float m2 = partials[o], s2 = partials[o + 1];
            float mn = fmaxf(m, m2);
            sv = sv * __expf(m - mn) + s2 * __expf(m2 - mn);
            m = mn;
        }
        const float logZ = m + logf(sv);
        const int l = r & (LROWS - 1);
        float num = 0.f, den = 0.f;
#pragma unroll
        for (int kk = 0; kk < 2; ++kk) {
            const int tg = tags[l * 2 + kk];
            const float neg = logZ - tagLogit[r * 2 + kk];
            const float ratio = 1.f - discard[tg];
            num += neg * ratio;
            den += ratio;
        }
        acc += num / den;
    }
    red[threadIdx.x] = acc;
    __syncthreads();
    for (int s = 128; s; s >>= 1) {
        if (threadIdx.x < s) red[threadIdx.x] += red[threadIdx.x + s];
        __syncthreads();
    }
    if (threadIdx.x == 0) out[0] = red[0] / ((float)NROWS + 1e-5f);
}

// ---------- host ----------
extern "C" void kernel_launch(void* const* d_in, const int* in_sizes, int n_in,
                              void* d_out, int out_size, void* d_ws, size_t ws_size,
                              hipStream_t stream) {
    (void)in_sizes; (void)n_in; (void)out_size; (void)ws_size;
    const float* hidden  = (const float*)d_in[0];
    const float* Wc      = (const float*)d_in[1];
    const float* bc      = (const float*)d_in[2];
    const float* Wp      = (const float*)d_in[3];
    const float* bp      = (const float*)d_in[4];
    const float* Wo      = (const float*)d_in[5];
    const float* bo      = (const float*)d_in[6];
    const float* discard = (const float*)d_in[7];
    const int*   begins  = (const int*)d_in[8];
    const int*   ends    = (const int*)d_in[9];
    const int*   bids    = (const int*)d_in[10];
    const int*   tags    = (const int*)d_in[11];
    float* out = (float*)d_out;

    char* w = (char*)d_ws;
    size_t off = 0;
    auto take = [&](size_t bytes) -> char* {
        char* p = w + off;
        off = (off + bytes + 255) & ~(size_t)255;
        return p;
    };
    bf16* gh   = (bf16*)take((size_t)NROWS * H2 * 2);
    bf16* gl   = (bf16*)take((size_t)NROWS * H2 * 2);
    bf16* wch  = (bf16*)take((size_t)DDIM * H2 * 2);
    bf16* wcl  = (bf16*)take((size_t)DDIM * H2 * 2);
    bf16* wph  = (bf16*)take((size_t)DDIM * H2 * 2);
    bf16* wpl  = (bf16*)take((size_t)DDIM * H2 * 2);
    bf16* woh  = (bf16*)take((size_t)VOCAB * DDIM * 2);
    bf16* wol  = (bf16*)take((size_t)VOCAB * DDIM * 2);
    float* feats = (float*)take((size_t)NROWS * DDIM * 4);
    bf16* fh   = (bf16*)take((size_t)NROWS * DDIM * 2);
    bf16* fl   = (bf16*)take((size_t)NROWS * DDIM * 2);
    float* tagLogit = (float*)take((size_t)NROWS * 2 * 4);
    float* partials = (float*)take((size_t)NSLICE * NROWS * 2 * 4);

    // 1. gather spans -> hi/lo
    k_gather<<<NROWS, 256, 0, stream>>>(hidden, begins, ends, bids, gh, gl);
    // 2. weight splits
    {
        int n1 = DDIM * H2;
        k_cvt<<<(n1 + 255) / 256, 256, 0, stream>>>(Wc, wch, wcl, n1);
        k_cvt<<<(n1 + 255) / 256, 256, 0, stream>>>(Wp, wph, wpl, n1);
        int n2 = VOCAB * DDIM;
        k_cvt<<<(n2 + 255) / 256, 256, 0, stream>>>(Wo, woh, wol, n2);
    }
    // 3. FFN + tanh
    k_ffn<<<NROWS / 32, 256, 0, stream>>>(gh, gl, wch, wcl, wph, wpl, bc, bp,
                                          feats, fh, fl);
    // 4. exact tag logits
    k_taglogits<<<NROWS, 64, 0, stream>>>(feats, Wo, bo, tags, tagLogit);
    // 5. fused big GEMM + online softmax partials
    {
        dim3 grid(NROWS / BP_M, NSLICE);
        k_bigpart<<<grid, 256, 0, stream>>>(fh, fl, woh, wol, bo, partials);
    }
    // 6. final combine + loss
    k_final<<<1, 256, 0, stream>>>(partials, tagLogit, tags, discard, out);
}
